// MSA_Outer_Product_Mean_34660386079047
// MI455X (gfx1250) — matrix operand order repaired and checked by reading, not compiled, not run on hardware
//
#include <hip/hip_runtime.h>
#include <hip/hip_bf16.h>

typedef __attribute__((ext_vector_type(8)))  _Float16 v8h;
typedef __attribute__((ext_vector_type(16))) _Float16 v16h;
typedef __attribute__((ext_vector_type(8)))  float    v8f;

// Problem constants (B=1), from setup_inputs(): R=64, L=256, D=256, H=32, P=128
#define RSEQ 64
#define LSEQ 256
#define DDIM 256
#define HDIM 32
#define PDIM 128
#define KHH  (HDIM * HDIM)        // 1024
#define OPM_LDK (KHH + 8)         // padded LDS stride (halves) to avoid bank conflicts

// ---------------------------------------------------------------------------
// WMMA operand loaders for v_wmma_f32_16x16x32_f16 (wave32).
// A matrix (MxK = 16x32, 16-bit), stored row-major [M][K]:
//   lane l (m = l&15, sel = l>>4): halves 0..7  = A[m][k0 + sel*8 + 0..7]
//                                  halves 8..15 = A[m][k0 + 16 + sel*8 + 0..7]
// B matrix (KxN = 32x16, 16-bit), stored column-major [N][K]:
//   lane l (n = l&15, sel = l>>4): halves 0..7  = B[k0 + sel*8 + 0..7][n]
//                                  halves 8..15 = B[k0 + 16 + sel*8 + 0..7][n]
// ---------------------------------------------------------------------------
__device__ __forceinline__ v16h load_wmma_a(const _Float16* base, int row, int ldk,
                                            int k0, int lane) {
  const _Float16* p = base + (size_t)row * ldk + k0 + ((lane >> 4) << 3);
  v8h lo = *reinterpret_cast<const v8h*>(p);
  v8h hi = *reinterpret_cast<const v8h*>(p + 16);
  return __builtin_shufflevector(lo, hi, 0, 1, 2, 3, 4, 5, 6, 7,
                                         8, 9, 10, 11, 12, 13, 14, 15);
}

__device__ __forceinline__ v16h load_wmma_b(const _Float16* base, int col, int ldk,
                                            int k0, int lane) {
  const _Float16* p = base + (size_t)col * ldk + k0 + ((lane >> 4) << 3);
  v8h lo = *reinterpret_cast<const v8h*>(p);
  v8h hi = *reinterpret_cast<const v8h*>(p + 16);
  return __builtin_shufflevector(lo, hi, 0, 1, 2, 3, 4, 5, 6, 7,
                                         8, 9, 10, 11, 12, 13, 14, 15);
}

// ---------------------------------------------------------------------------
// Kernel 0: convert w3 [P][1024] fp32 -> f16
// ---------------------------------------------------------------------------
__global__ __launch_bounds__(256) void cvt_w3_kernel(const float* __restrict__ w3,
                                                     _Float16* __restrict__ w3h) {
  int idx = blockIdx.x * 256 + threadIdx.x;
  w3h[idx] = (_Float16)w3[idx];
}

// ---------------------------------------------------------------------------
// Kernel 1: LayerNorm + both projections. One wave32 per (r,l) row.
// Outputs transposed f16 tensors:
//   a1t[l][h][r] = msa_1 = LN(x)@w1^T + b1   (the 'y' / j axis)
//   a2t[l][h][r] = msa_2 = LN(x)@w2^T + b2   (the 'x' / i axis)
// ---------------------------------------------------------------------------
__global__ __launch_bounds__(256) void ln_proj_kernel(
    const float* __restrict__ msa, const float* __restrict__ gamma,
    const float* __restrict__ beta, const float* __restrict__ w1,
    const float* __restrict__ b1, const float* __restrict__ w2,
    const float* __restrict__ b2, _Float16* __restrict__ a1t,
    _Float16* __restrict__ a2t) {
  __shared__ float xs[8][264];  // 8 waves * padded row of normalized x
  const int lane = threadIdx.x & 31;
  const int wave = threadIdx.x >> 5;
  const int row  = blockIdx.x * 8 + wave;  // 0 .. R*L-1
  const int l    = row & (LSEQ - 1);
  const int r    = row >> 8;               // log2(LSEQ) == 8

  const float* xp = msa + (size_t)row * DDIM + lane * 8;
  float4 v0 = *reinterpret_cast<const float4*>(xp);
  float4 v1 = *reinterpret_cast<const float4*>(xp + 4);

  float s  = v0.x + v0.y + v0.z + v0.w + v1.x + v1.y + v1.z + v1.w;
  float ss = v0.x * v0.x + v0.y * v0.y + v0.z * v0.z + v0.w * v0.w +
             v1.x * v1.x + v1.y * v1.y + v1.z * v1.z + v1.w * v1.w;
  #pragma unroll
  for (int off = 16; off > 0; off >>= 1) {
    s  += __shfl_xor(s, off, 32);
    ss += __shfl_xor(ss, off, 32);
  }
  const float mu   = s * (1.0f / DDIM);
  const float var  = ss * (1.0f / DDIM) - mu * mu;
  const float rstd = rsqrtf(var + 1e-5f);

  const float4 g0 = *reinterpret_cast<const float4*>(gamma + lane * 8);
  const float4 g1 = *reinterpret_cast<const float4*>(gamma + lane * 8 + 4);
  const float4 e0 = *reinterpret_cast<const float4*>(beta + lane * 8);
  const float4 e1 = *reinterpret_cast<const float4*>(beta + lane * 8 + 4);
  float4 n0, n1;
  n0.x = (v0.x - mu) * rstd * g0.x + e0.x;
  n0.y = (v0.y - mu) * rstd * g0.y + e0.y;
  n0.z = (v0.z - mu) * rstd * g0.z + e0.z;
  n0.w = (v0.w - mu) * rstd * g0.w + e0.w;
  n1.x = (v1.x - mu) * rstd * g1.x + e1.x;
  n1.y = (v1.y - mu) * rstd * g1.y + e1.y;
  n1.z = (v1.z - mu) * rstd * g1.z + e1.z;
  n1.w = (v1.w - mu) * rstd * g1.w + e1.w;
  *reinterpret_cast<float4*>(&xs[wave][lane * 8])     = n0;
  *reinterpret_cast<float4*>(&xs[wave][lane * 8 + 4]) = n1;
  __syncthreads();

  // Each lane computes output channel h = lane for both projections.
  const int h = lane;
  const float* w1r = w1 + (size_t)h * DDIM;
  const float* w2r = w2 + (size_t)h * DDIM;
  float acc1 = b1[h];
  float acc2 = b2[h];
  #pragma unroll 8
  for (int d = 0; d < DDIM; d += 4) {
    float4 xv = *reinterpret_cast<const float4*>(&xs[wave][d]);   // LDS broadcast
    float4 wa = *reinterpret_cast<const float4*>(w1r + d);
    float4 wb = *reinterpret_cast<const float4*>(w2r + d);
    acc1 += xv.x * wa.x + xv.y * wa.y + xv.z * wa.z + xv.w * wa.w;
    acc2 += xv.x * wb.x + xv.y * wb.y + xv.z * wb.z + xv.w * wb.w;
  }
  const size_t oidx = ((size_t)l * HDIM + h) * RSEQ + r;
  a1t[oidx] = (_Float16)acc1;
  a2t[oidx] = (_Float16)acc2;
}

// ---------------------------------------------------------------------------
// Kernel 2 (fused): per block, a 4x4 tile of (i,j) pairs.
// Stage B: opm[pair][x][y] = (1/R) * a2t[i][x][:] . a1t[j][y][:]   (K = 64)
//          via v_wmma_f32_16x16x32_f16, stored as f16 into LDS in stage-C
//          B-operand layout ([pair][q=x*32+y], q contiguous, padded stride).
// Stage C: out[i][j][p] = w3h[p][:] . opm[pair][:] + b3[p]          (K = 1024)
//          each wave owns one 16-row p-tile, N = all 16 pairs.
// ---------------------------------------------------------------------------
__global__ __launch_bounds__(256) void opm_fused_kernel(
    const _Float16* __restrict__ a2t,  // [L][H][R]  (i, x, r)
    const _Float16* __restrict__ a1t,  // [L][H][R]  (j, y, r)
    const _Float16* __restrict__ w3h,  // [P][1024]
    const float* __restrict__ b3,      // [P]
    float* __restrict__ out) {         // [L][L][P]
  __shared__ _Float16 opm_s[16 * OPM_LDK];  // 16 pairs * (1024 + 8) halves = 33 KB

  const int lane = threadIdx.x & 31;
  const int wave = threadIdx.x >> 5;
  const int m    = lane & 15;
  const int sel  = lane >> 4;
  const int i0   = blockIdx.y * 4;
  const int j0   = blockIdx.x * 4;
  const float invR = 1.0f / (float)RSEQ;

  // ---- Stage B: each wave computes 2 of the 16 pairs -----------------------
  #pragma unroll
  for (int pp = 0; pp < 2; ++pp) {
    const int pair = wave * 2 + pp;
    const int i = i0 + (pair >> 2);
    const int j = j0 + (pair & 3);
    const _Float16* Abase = a2t + (size_t)i * (HDIM * RSEQ);
    const _Float16* Bbase = a1t + (size_t)j * (HDIM * RSEQ);
    #pragma unroll
    for (int xt = 0; xt < 2; ++xt) {
      #pragma unroll
      for (int yt = 0; yt < 2; ++yt) {
        v8f acc = {};
        #pragma unroll
        for (int k0 = 0; k0 < RSEQ; k0 += 32) {
          v16h a = load_wmma_a(Abase, xt * 16 + m, RSEQ, k0, lane);
          v16h b = load_wmma_b(Bbase, yt * 16 + m, RSEQ, k0, lane);
          acc = __builtin_amdgcn_wmma_f32_16x16x32_f16(
              false, a, false, b, (short)0, acc, false, false);
        }
        // D layout: lane = N(y), vgpr k -> M(x) = k + 8*sel
        #pragma unroll
        for (int k = 0; k < 8; ++k) {
          const int x = xt * 16 + k + 8 * sel;
          const int y = yt * 16 + m;
          opm_s[pair * OPM_LDK + x * HDIM + y] = (_Float16)(acc[k] * invR);
        }
      }
    }
  }
  __syncthreads();

  // ---- Stage C: wave owns p-tile [p0, p0+16), N = 16 pairs, K = 1024 -------
  const int p0 = wave * 16;
  v8f acc = {};
  #pragma unroll 4
  for (int k0 = 0; k0 < KHH; k0 += 32) {
    v16h a = load_wmma_a(w3h, p0 + m, KHH, k0, lane);        // A: w3h[p][q]
    v16h b = load_wmma_b(opm_s, m, OPM_LDK, k0, lane);       // B: opm[q][pair]
    acc = __builtin_amdgcn_wmma_f32_16x16x32_f16(
        false, a, false, b, (short)0, acc, false, false);
  }
  #pragma unroll
  for (int k = 0; k < 8; ++k) {
    const int p    = p0 + k + 8 * sel;
    const int pair = m;
    const int i = i0 + (pair >> 2);
    const int j = j0 + (pair & 3);
    out[((size_t)i * LSEQ + j) * PDIM + p] = acc[k] + b3[p];
  }
}

// ---------------------------------------------------------------------------
extern "C" void kernel_launch(void* const* d_in, const int* in_sizes, int n_in,
                              void* d_out, int out_size, void* d_ws, size_t ws_size,
                              hipStream_t stream) {
  const float* msa   = (const float*)d_in[0];
  const float* gamma = (const float*)d_in[1];
  const float* beta  = (const float*)d_in[2];
  const float* w1    = (const float*)d_in[3];
  const float* b1    = (const float*)d_in[4];
  const float* w2    = (const float*)d_in[5];
  const float* b2    = (const float*)d_in[6];
  const float* w3    = (const float*)d_in[7];
  const float* b3    = (const float*)d_in[8];
  float* out = (float*)d_out;

  char* ws = (char*)d_ws;
  _Float16* a1t = (_Float16*)(ws);                    // L*H*R halves = 1 MB
  _Float16* a2t = (_Float16*)(ws + (1u << 20));       // 1 MB
  _Float16* w3h = (_Float16*)(ws + (2u << 20));       // P*1024 halves = 256 KB

  // w3 fp32 -> f16
  cvt_w3_kernel<<<(PDIM * KHH) / 256, 256, 0, stream>>>(w3, w3h);
  // LayerNorm + projections: R*L = 16384 rows, 8 waves/block
  ln_proj_kernel<<<(RSEQ * LSEQ) / 8, 256, 0, stream>>>(
      msa, gamma, beta, w1, b1, w2, b2, a1t, a2t);
  // Fused outer-product-mean + pair projection: (L/4) x (L/4) blocks
  opm_fused_kernel<<<dim3(LSEQ / 4, LSEQ / 4), 256, 0, stream>>>(
      a2t, a1t, w3h, b3, out);
}
